// Block_21148418966360
// MI455X (gfx1250) — compile-verified, matrix operand order checked by reading, not run on hardware
//
#include <hip/hip_runtime.h>
#include <hip/hip_bf16.h>

// ---------------------------------------------------------------------------
// GPT-2 block for MI455X (gfx1250, wave32, WMMA).
// All GEMMs run through V_WMMA_F32_16X16X32_BF16 (f32 accumulate).
// Attention is a flash-style streaming kernel (QK^T and PV both via WMMA).
// Global->LDS staging uses CDNA5 async LDS copies (ASYNCcnt) with LDS
// double-buffering when the toolchain exposes the builtins.
// Workspace requirement: ~330 MB (see layout in kernel_launch).
// ---------------------------------------------------------------------------

#define B_  4
#define S_  2048
#define H_  1024
#define NH_ 16
#define D_  64
#define BS_ (B_ * S_)

typedef __attribute__((ext_vector_type(16))) __bf16 v16bf;
typedef __attribute__((ext_vector_type(8)))  float  v8f;
typedef unsigned int u32x4 __attribute__((ext_vector_type(4)));
typedef unsigned int u32x2 __attribute__((ext_vector_type(2)));
typedef int v4i_gl __attribute__((vector_size(16)));   // builtin's pointee type

union AFrag { v16bf v; unsigned short u[16]; u32x4 q[2]; };
union CFrag { v8f   v; float f[8]; };

// ---- CDNA5 async global->LDS copy (ASYNCcnt) -------------------------------
#if defined(__has_builtin)
# if __has_builtin(__builtin_amdgcn_global_load_async_to_lds_b128)
#  define HAVE_ASYNC_COPY 1
# endif
#endif

__device__ __forceinline__ void async_copy_b128(const void* g, void* l) {
#ifdef HAVE_ASYNC_COPY
    __builtin_amdgcn_global_load_async_to_lds_b128(
        (__attribute__((address_space(1))) v4i_gl*)(void*)g,
        (__attribute__((address_space(3))) v4i_gl*)l, 0, 0);
#else
    (void)g; (void)l;
#endif
}

__device__ __forceinline__ void async_wait0() {
#ifdef HAVE_ASYNC_COPY
# if __has_builtin(__builtin_amdgcn_s_wait_asynccnt)
    __builtin_amdgcn_s_wait_asynccnt(0);
# else
    asm volatile("s_wait_asynccnt 0x0" ::: "memory");
# endif
#endif
}

__device__ __forceinline__ unsigned short f2bf(float f) {
    unsigned int u = __float_as_uint(f);
    u += 0x7FFFu + ((u >> 16) & 1u);          // round-to-nearest-even
    return (unsigned short)(u >> 16);
}

__device__ __forceinline__ float gelu_f(float x) {
    float x3 = x * x * x;
    return 0.5f * x * (1.0f + tanhf(0.7978845608028654f * (x + 0.044715f * x3)));
}

// ---------------------------------------------------------------------------
// Weight convert + transpose: W f32 [K][N]  ->  Wt bf16 [N][K]
// (so GEMM B-tiles are contiguous [n][k], matching the wave32 B-fragment).
// K, N multiples of 32. Block (32,8).
// ---------------------------------------------------------------------------
__global__ void transpose_f32_to_bf16(const float* __restrict__ W,
                                      unsigned short* __restrict__ Wt,
                                      int K, int N) {
    __shared__ float t[32][33];
    int k0 = blockIdx.y << 5, n0 = blockIdx.x << 5;
    int tx = threadIdx.x, ty = threadIdx.y;
#pragma unroll
    for (int i = 0; i < 32; i += 8)
        t[ty + i][tx] = W[(size_t)(k0 + ty + i) * N + n0 + tx];
    __syncthreads();
#pragma unroll
    for (int i = 0; i < 32; i += 8)
        Wt[(size_t)(n0 + ty + i) * K + k0 + tx] = f2bf(t[tx][ty + i]);
}

// ---------------------------------------------------------------------------
// LayerNorm over H=1024, output bf16.  One block (256 thr) per row.
// ---------------------------------------------------------------------------
__global__ void layernorm_bf16_kernel(const float* __restrict__ x,
                                      const float* __restrict__ g,
                                      const float* __restrict__ bb,
                                      unsigned short* __restrict__ out) {
    const int Hc = H_;
    int row = blockIdx.x;
    int tid = threadIdx.x;
    const float* xr = x + (size_t)row * Hc;
    float v[4];
    float s = 0.f;
#pragma unroll
    for (int i = 0; i < 4; ++i) { v[i] = xr[tid + i * 256]; s += v[i]; }
    __shared__ float red[256];
    red[tid] = s; __syncthreads();
    for (int off = 128; off > 0; off >>= 1) {
        if (tid < off) red[tid] += red[tid + off];
        __syncthreads();
    }
    float mean = red[0] * (1.0f / (float)Hc);
    __syncthreads();
    float s2 = 0.f;
#pragma unroll
    for (int i = 0; i < 4; ++i) { float d = v[i] - mean; s2 += d * d; }
    red[tid] = s2; __syncthreads();
    for (int off = 128; off > 0; off >>= 1) {
        if (tid < off) red[tid] += red[tid + off];
        __syncthreads();
    }
    float rstd = rsqrtf(red[0] * (1.0f / (float)Hc) + 1e-5f);
#pragma unroll
    for (int i = 0; i < 4; ++i) {
        int col = tid + i * 256;
        float y = (v[i] - mean) * rstd * g[col] + bb[col];
        out[(size_t)row * Hc + col] = f2bf(y);
    }
}

// ---------------------------------------------------------------------------
// WMMA bf16 GEMM:  C[M][N] = A[M][K] * Bt[N][K]^T + bias  (+gelu)(+residual)
// Block: 256 thr = 8 waves (2 M x 4 N), each wave 64x32 out (4x2 WMMA tiles).
// Block tile 128x128, K-step 32, double-buffered async LDS staging.
// M%128==0, N%128==0, K%32==0 (true for all call sites here).
// ---------------------------------------------------------------------------
#define GTM 128
#define GTN 128
#define GTK 32
#define LDS_STRIDE 40   // 32 + 8 pad (ushorts); row pitch 80 B (16 B aligned)

__global__ __launch_bounds__(256)
void gemm_bf16_wmma(const unsigned short* __restrict__ A,
                    const unsigned short* __restrict__ Bt,
                    const float* __restrict__ bias,
                    const float* __restrict__ residual,  // nullable, [M][N]
                    float* __restrict__ outF,            // nullable
                    unsigned short* __restrict__ outB,   // nullable
                    int M, int N, int K, int do_gelu) {
    __shared__ alignas(16) unsigned short sA[2][GTM * LDS_STRIDE];
    __shared__ alignas(16) unsigned short sB[2][GTN * LDS_STRIDE];

    const int tid  = threadIdx.x;
    const int bm   = blockIdx.y * GTM;
    const int bn   = blockIdx.x * GTN;
    const int lane = tid & 31;
    const int wave = tid >> 5;
    const int wm   = (wave & 1) * 64;   // wave M offset in tile
    const int wn   = (wave >> 1) * 32;  // wave N offset in tile
    const int lr   = lane & 15;
    const int kb   = (lane >= 16) ? 8 : 0;
    const int hi8  = (lane >= 16) ? 8 : 0;

    CFrag acc[4][2];
#pragma unroll
    for (int mi = 0; mi < 4; ++mi)
#pragma unroll
        for (int ni = 0; ni < 2; ++ni)
#pragma unroll
            for (int r = 0; r < 8; ++r) acc[mi][ni].f[r] = 0.f;

    auto fill = [&](int buf, int kt) {
#ifdef HAVE_ASYNC_COPY
        // 512 chunks of 8 ushorts (16 B) per tile; 256 threads -> 2 each.
#pragma unroll
        for (int i = 0; i < 2; ++i) {
            int idx = tid + i * 256;
            int row = idx >> 2;
            int c8  = (idx & 3) * 8;
            async_copy_b128(&A[(size_t)(bm + row) * K + kt + c8],
                            &sA[buf][row * LDS_STRIDE + c8]);
            async_copy_b128(&Bt[(size_t)(bn + row) * K + kt + c8],
                            &sB[buf][row * LDS_STRIDE + c8]);
        }
#else
        // fallback: VGPR round-trip, 4 chunks of 4 ushorts per thread per tile
#pragma unroll
        for (int i = 0; i < 4; ++i) {
            int idx = tid + i * 256;
            int row = idx >> 3;
            int c4  = (idx & 7) * 4;
            *(u32x2*)&sA[buf][row * LDS_STRIDE + c4] =
                *(const u32x2*)&A[(size_t)(bm + row) * K + kt + c4];
            *(u32x2*)&sB[buf][row * LDS_STRIDE + c4] =
                *(const u32x2*)&Bt[(size_t)(bn + row) * K + kt + c4];
        }
#endif
    };

    fill(0, 0);
    async_wait0();
    __syncthreads();

    const int nT = K / GTK;
    for (int t = 0; t < nT; ++t) {
        const int cur = t & 1;
        if (t + 1 < nT) fill(cur ^ 1, (t + 1) * GTK);   // overlap with compute
        if (t + 2 < nT) {                               // warm L2 for t+2
            __builtin_prefetch(&A[(size_t)(bm + (tid >> 1)) * K + (t + 2) * GTK], 0, 3);
            __builtin_prefetch(&Bt[(size_t)(bn + (tid >> 1)) * K + (t + 2) * GTK], 0, 3);
        }

        AFrag a[4], b[2];
#pragma unroll
        for (int mi = 0; mi < 4; ++mi) {
            const unsigned short* p = &sA[cur][(wm + mi * 16 + lr) * LDS_STRIDE + kb];
            a[mi].q[0] = *(const u32x4*)p;
            a[mi].q[1] = *(const u32x4*)(p + 16);
        }
#pragma unroll
        for (int ni = 0; ni < 2; ++ni) {
            const unsigned short* p = &sB[cur][(wn + ni * 16 + lr) * LDS_STRIDE + kb];
            b[ni].q[0] = *(const u32x4*)p;
            b[ni].q[1] = *(const u32x4*)(p + 16);
        }
#pragma unroll
        for (int mi = 0; mi < 4; ++mi)
#pragma unroll
            for (int ni = 0; ni < 2; ++ni)
                acc[mi][ni].v = __builtin_amdgcn_wmma_f32_16x16x32_bf16(
                    false, a[mi].v, false, b[ni].v,
                    (short)0, acc[mi][ni].v, false, false);

        if (t + 1 < nT) { async_wait0(); __syncthreads(); }
    }

    // epilogue: bias (+gelu) (+residual) -> f32 and/or bf16
#pragma unroll
    for (int mi = 0; mi < 4; ++mi) {
#pragma unroll
        for (int ni = 0; ni < 2; ++ni) {
            int rowBase = bm + wm + mi * 16 + hi8;
            int col     = bn + wn + ni * 16 + lr;
            float bsv   = bias[col];
#pragma unroll
            for (int r = 0; r < 8; ++r) {
                int row = rowBase + r;
                float v = acc[mi][ni].f[r] + bsv;
                if (do_gelu) v = gelu_f(v);
                if (residual) v += residual[(size_t)row * N + col];
                if (outF) outF[(size_t)row * N + col] = v;
                if (outB) outB[(size_t)row * N + col] = f2bf(v);
            }
        }
    }
}

// ---------------------------------------------------------------------------
// Split qkv f32 [BS][3H] -> Q/K/V bf16 planes [B*NH][S][D] + present f32
// present: [B][2][NH][S][D]
// ---------------------------------------------------------------------------
__global__ void qkv_pack_kernel(const float* __restrict__ qkv,
                                unsigned short* __restrict__ Qb,
                                unsigned short* __restrict__ Kb,
                                unsigned short* __restrict__ Vb,
                                float* __restrict__ present) {
    size_t e = (size_t)blockIdx.x * blockDim.x + threadIdx.x;
    const size_t total = (size_t)BS_ * 3 * H_;
    if (e >= total) return;
    int col = (int)(e % (3 * H_));
    int bs  = (int)(e / (3 * H_));
    int b   = bs / S_;
    int s   = bs % S_;
    int which = col / H_;
    int hcol  = col % H_;
    int h = hcol / D_;
    int d = hcol % D_;
    float val = qkv[e];
    unsigned short bv = f2bf(val);
    size_t plane = ((size_t)(b * NH_ + h) * S_ + s) * D_ + d;
    if (which == 0) {
        Qb[plane] = bv;
    } else if (which == 1) {
        Kb[plane] = bv;
        present[(((size_t)(b * 2 + 0) * NH_ + h) * S_ + s) * D_ + d] = val;
    } else {
        Vb[plane] = bv;
        present[(((size_t)(b * 2 + 1) * NH_ + h) * S_ + s) * D_ + d] = val;
    }
}

// ---------------------------------------------------------------------------
// Flash attention (causal).  grid = (S/64, B*NH), block = 128 (4 waves).
// Each wave owns 16 q rows; streams 64-key tiles; online softmax.
// QK^T and PV both via WMMA bf16.  Output attnout bf16 [B][S][H].
// ---------------------------------------------------------------------------
#define FS 80   // LDS row stride (ushorts): 64 + 16 pad, pitch 160 B

__global__ __launch_bounds__(128)
void flash_attn_kernel(const unsigned short* __restrict__ Qb,
                       const unsigned short* __restrict__ Kb,
                       const unsigned short* __restrict__ Vb,
                       unsigned short* __restrict__ attnout) {
    __shared__ alignas(16) unsigned short sQ[64 * FS];
    __shared__ alignas(16) unsigned short sK[64 * FS];
    __shared__ alignas(16) unsigned short sVt[64 * FS];   // [d][key]
    __shared__ alignas(16) unsigned short sP[4 * 16 * FS];

    const int qt  = blockIdx.x;          // q tile (64 rows)
    const int hb  = blockIdx.y;          // b*NH + h
    const int b   = hb / NH_;
    const int h   = hb % NH_;
    const int tid  = threadIdx.x;
    const int lane = tid & 31;
    const int w    = tid >> 5;
    const int lr   = lane & 15;
    const int kb   = (lane >= 16) ? 8 : 0;
    const int hi8  = (lane >= 16) ? 8 : 0;
    const float scale = 0.125f;          // 1/sqrt(64)

    const unsigned short* Qp  = Qb + ((size_t)hb * S_ + qt * 64) * D_;
    const unsigned short* Kp0 = Kb + (size_t)hb * S_ * D_;
    const unsigned short* Vp0 = Vb + (size_t)hb * S_ * D_;

    // stage Q tile (64x64): 512 chunks of 8 ushorts, 128 threads -> 4 each
#ifdef HAVE_ASYNC_COPY
#pragma unroll
    for (int i = 0; i < 4; ++i) {
        int c   = tid + i * 128;
        int row = c >> 3;
        int d8  = (c & 7) * 8;
        async_copy_b128(&Qp[row * D_ + d8], &sQ[row * FS + d8]);
    }
    async_wait0();
#else
#pragma unroll
    for (int i = 0; i < 8; ++i) {
        int c   = tid + i * 128;
        int row = c >> 4;
        int d4  = (c & 15) * 4;
        *(u32x2*)&sQ[row * FS + d4] = *(const u32x2*)&Qp[row * D_ + d4];
    }
#endif
    __syncthreads();

    AFrag qa[2];
#pragma unroll
    for (int kk = 0; kk < 2; ++kk) {
        const unsigned short* p = &sQ[(w * 16 + lr) * FS + kk * 32 + kb];
        qa[kk].q[0] = *(const u32x4*)p;
        qa[kk].q[1] = *(const u32x4*)(p + 16);
    }

    float m[8], l[8];
    CFrag o[4];
#pragma unroll
    for (int r = 0; r < 8; ++r) { m[r] = -1e30f; l[r] = 0.f; }
#pragma unroll
    for (int di = 0; di < 4; ++di)
#pragma unroll
        for (int r = 0; r < 8; ++r) o[di].f[r] = 0.f;

    for (int kt = 0; kt <= qt; ++kt) {
        const unsigned short* Kp = Kp0 + (size_t)kt * 64 * D_;
        const unsigned short* Vp = Vp0 + (size_t)kt * 64 * D_;
        // stage K [key][d] (async) and V^T [d][key] (VGPR transpose)
#ifdef HAVE_ASYNC_COPY
#pragma unroll
        for (int i = 0; i < 4; ++i) {
            int c   = tid + i * 128;
            int row = c >> 3;
            int d8  = (c & 7) * 8;
            async_copy_b128(&Kp[row * D_ + d8], &sK[row * FS + d8]);
        }
#else
#pragma unroll
        for (int i = 0; i < 8; ++i) {
            int c   = tid + i * 128;
            int row = c >> 4;
            int d4  = (c & 15) * 4;
            *(u32x2*)&sK[row * FS + d4] = *(const u32x2*)&Kp[row * D_ + d4];
        }
#endif
#pragma unroll
        for (int i = 0; i < 32; ++i) {
            int idx = tid + i * 128;
            int d   = idx & 63;
            int key = idx >> 6;
            sVt[d * FS + key] = Vp[key * D_ + d];
        }
        async_wait0();
        __syncthreads();

        // scores = Q * K^T  (16 q rows x 64 keys per wave)
        CFrag sc[4];
#pragma unroll
        for (int ni = 0; ni < 4; ++ni) {
#pragma unroll
            for (int r = 0; r < 8; ++r) sc[ni].f[r] = 0.f;
#pragma unroll
            for (int kk = 0; kk < 2; ++kk) {
                AFrag bk;
                const unsigned short* p = &sK[(ni * 16 + lr) * FS + kk * 32 + kb];
                bk.q[0] = *(const u32x4*)p;
                bk.q[1] = *(const u32x4*)(p + 16);
                sc[ni].v = __builtin_amdgcn_wmma_f32_16x16x32_bf16(
                    false, qa[kk].v, false, bk.v, (short)0, sc[ni].v, false, false);
            }
        }

        const bool diag = (kt == qt);
        // online softmax per row; write P (bf16) to LDS for re-layout
#pragma unroll
        for (int r = 0; r < 8; ++r) {
            int qrow = qt * 64 + w * 16 + r + hi8;
            float pv[4];
            float tmax = -1e30f;
#pragma unroll
            for (int ni = 0; ni < 4; ++ni) {
                float v = sc[ni].f[r] * scale;
                if (diag) {
                    int kcol = kt * 64 + ni * 16 + lr;
                    if (kcol > qrow) v = -1e30f;
                }
                pv[ni] = v;
                tmax = fmaxf(tmax, v);
            }
#pragma unroll
            for (int msk = 1; msk < 16; msk <<= 1)
                tmax = fmaxf(tmax, __shfl_xor(tmax, msk, 32));
            float newm  = fmaxf(m[r], tmax);
            float alpha = __expf(m[r] - newm);
            float rsum  = 0.f;
#pragma unroll
            for (int ni = 0; ni < 4; ++ni) {
                float p = __expf(pv[ni] - newm);
                pv[ni] = p;
                rsum += p;
            }
#pragma unroll
            for (int msk = 1; msk < 16; msk <<= 1)
                rsum += __shfl_xor(rsum, msk, 32);
            m[r] = newm;
            l[r] = l[r] * alpha + rsum;
#pragma unroll
            for (int di = 0; di < 4; ++di) o[di].f[r] *= alpha;
#pragma unroll
            for (int ni = 0; ni < 4; ++ni)
                sP[(w * 16 + r + hi8) * FS + ni * 16 + lr] = f2bf(pv[ni]);
        }
        __syncthreads();

        // O += P * V   (A = P 16x64, Bt = V^T [d][key])
        AFrag pa[2];
#pragma unroll
        for (int kk = 0; kk < 2; ++kk) {
            const unsigned short* p = &sP[(w * 16 + lr) * FS + kk * 32 + kb];
            pa[kk].q[0] = *(const u32x4*)p;
            pa[kk].q[1] = *(const u32x4*)(p + 16);
        }
#pragma unroll
        for (int di = 0; di < 4; ++di) {
#pragma unroll
            for (int kk = 0; kk < 2; ++kk) {
                AFrag bv;
                const unsigned short* p = &sVt[(di * 16 + lr) * FS + kk * 32 + kb];
                bv.q[0] = *(const u32x4*)p;
                bv.q[1] = *(const u32x4*)(p + 16);
                o[di].v = __builtin_amdgcn_wmma_f32_16x16x32_bf16(
                    false, pa[kk].v, false, bv.v, (short)0, o[di].v, false, false);
            }
        }
        __syncthreads();
    }

    // normalize + store attnout[b][s][h*64+d]  (bf16)
#pragma unroll
    for (int di = 0; di < 4; ++di) {
#pragma unroll
        for (int r = 0; r < 8; ++r) {
            int srow = qt * 64 + w * 16 + r + hi8;
            int d    = di * 16 + lr;
            float v  = o[di].f[r] * (1.0f / l[r]);
            attnout[((size_t)b * S_ + srow) * H_ + h * D_ + d] = f2bf(v);
        }
    }
}

// ---------------------------------------------------------------------------
// Host-side orchestration
// ---------------------------------------------------------------------------
extern "C" void kernel_launch(void* const* d_in, const int* in_sizes, int n_in,
                              void* d_out, int out_size, void* d_ws, size_t ws_size,
                              hipStream_t stream) {
    (void)in_sizes; (void)n_in; (void)out_size; (void)ws_size;
    const float* x       = (const float*)d_in[0];
    const float* ln1_g   = (const float*)d_in[1];
    const float* ln1_b   = (const float*)d_in[2];
    const float* w_attn  = (const float*)d_in[3];
    const float* b_attn  = (const float*)d_in[4];
    const float* w_aproj = (const float*)d_in[5];
    const float* b_aproj = (const float*)d_in[6];
    const float* ln2_g   = (const float*)d_in[7];
    const float* ln2_b   = (const float*)d_in[8];
    const float* w_fc    = (const float*)d_in[9];
    const float* b_fc    = (const float*)d_in[10];
    const float* w_mproj = (const float*)d_in[11];
    const float* b_mproj = (const float*)d_in[12];

    float* out_x       = (float*)d_out;                      // [B,S,H]
    float* out_present = (float*)d_out + (size_t)BS_ * H_;   // [B,2,NH,S,D]

    // ---- workspace layout (bytes) ----
    char* ws = (char*)d_ws;
    size_t off = 0;
    auto alloc = [&](size_t bytes) { void* p = ws + off; off += (bytes + 255) & ~(size_t)255; return p; };
    unsigned short* h1      = (unsigned short*)alloc((size_t)BS_ * H_ * 2);
    unsigned short* wT_attn = (unsigned short*)alloc((size_t)3 * H_ * H_ * 2);
    unsigned short* wT_apr  = (unsigned short*)alloc((size_t)H_ * H_ * 2);
    unsigned short* wT_fc   = (unsigned short*)alloc((size_t)4 * H_ * H_ * 2);
    unsigned short* wT_mpr  = (unsigned short*)alloc((size_t)4 * H_ * H_ * 2);
    float*          qkv     = (float*)alloc((size_t)BS_ * 3 * H_ * 4);
    unsigned short* Qb      = (unsigned short*)alloc((size_t)B_ * NH_ * S_ * D_ * 2);
    unsigned short* Kb      = (unsigned short*)alloc((size_t)B_ * NH_ * S_ * D_ * 2);
    unsigned short* Vb      = (unsigned short*)alloc((size_t)B_ * NH_ * S_ * D_ * 2);
    unsigned short* attnout = (unsigned short*)alloc((size_t)BS_ * H_ * 2);
    float*          x2      = (float*)alloc((size_t)BS_ * H_ * 4);
    unsigned short* h2      = (unsigned short*)alloc((size_t)BS_ * H_ * 2);
    unsigned short* m1      = (unsigned short*)alloc((size_t)BS_ * 4 * H_ * 2);

    dim3 tb32(32, 8);
    // 1) weights -> bf16, transposed to [N][K]
    transpose_f32_to_bf16<<<dim3(3 * H_ / 32, H_ / 32), tb32, 0, stream>>>(w_attn,  wT_attn, H_,     3 * H_);
    transpose_f32_to_bf16<<<dim3(H_ / 32,     H_ / 32), tb32, 0, stream>>>(w_aproj, wT_apr,  H_,     H_);
    transpose_f32_to_bf16<<<dim3(4 * H_ / 32, H_ / 32), tb32, 0, stream>>>(w_fc,    wT_fc,   H_,     4 * H_);
    transpose_f32_to_bf16<<<dim3(H_ / 32, 4 * H_ / 32), tb32, 0, stream>>>(w_mproj, wT_mpr,  4 * H_, H_);

    // 2) ln1(x) -> h1 (bf16)
    layernorm_bf16_kernel<<<BS_, 256, 0, stream>>>(x, ln1_g, ln1_b, h1);

    // 3) qkv = h1 @ w_attn + b_attn  (f32 out)
    gemm_bf16_wmma<<<dim3(3 * H_ / GTN, BS_ / GTM), 256, 0, stream>>>(
        h1, wT_attn, b_attn, nullptr, qkv, nullptr, BS_, 3 * H_, H_, 0);

    // 4) split heads, write present
    {
        size_t total = (size_t)BS_ * 3 * H_;
        qkv_pack_kernel<<<(unsigned)((total + 255) / 256), 256, 0, stream>>>(
            qkv, Qb, Kb, Vb, out_present);
    }

    // 5) flash attention -> attnout (bf16, [B][S][H])
    flash_attn_kernel<<<dim3(S_ / 64, B_ * NH_), 128, 0, stream>>>(Qb, Kb, Vb, attnout);

    // 6) x2 = x + attnout @ w_aproj + b_aproj   (f32)
    gemm_bf16_wmma<<<dim3(H_ / GTN, BS_ / GTM), 256, 0, stream>>>(
        attnout, wT_apr, b_aproj, x, x2, nullptr, BS_, H_, H_, 0);

    // 7) ln2(x2) -> h2 (bf16)
    layernorm_bf16_kernel<<<BS_, 256, 0, stream>>>(x2, ln2_g, ln2_b, h2);

    // 8) m1 = gelu(h2 @ w_fc + b_fc)   (bf16)
    gemm_bf16_wmma<<<dim3(4 * H_ / GTN, BS_ / GTM), 256, 0, stream>>>(
        h2, wT_fc, b_fc, nullptr, nullptr, m1, BS_, 4 * H_, H_, 1);

    // 9) out_x = x2 + m1 @ w_mproj + b_mproj   (f32)
    gemm_bf16_wmma<<<dim3(H_ / GTN, BS_ / GTM), 256, 0, stream>>>(
        m1, wT_mpr, b_mproj, x2, out_x, nullptr, BS_, H_, 4 * H_, 0);
}
